// BoundaryExpert_64896955843136
// MI455X (gfx1250) — matrix operand order repaired and checked by reading, not compiled, not run on hardware
//
#include <hip/hip_runtime.h>

// ---------------- problem constants (fixed by the reference) ----------------
constexpr int kC    = 512;     // feat_dim
constexpr int kT    = 8192;    // sequence length
constexpr int kN    = 16384;   // proposals
constexpr int kHID  = 1024;
constexpr int kOUT  = 512;
constexpr int k2C   = 2 * kC;  // 1024

// ---------------- CDNA5 WMMA types ----------------
typedef __attribute__((ext_vector_type(16))) __bf16 v16bf;
typedef __attribute__((ext_vector_type(8)))  float  v8f;
typedef __attribute__((ext_vector_type(4)))  int    v4i;

static __device__ __forceinline__ __bf16 to_bf16(float f) {
  unsigned u = __builtin_bit_cast(unsigned, f);
  unsigned r = (u + 0x7FFFu + ((u >> 16) & 1u)) >> 16;
  unsigned short s = (unsigned short)r;
  return __builtin_bit_cast(__bf16, s);
}

// ---------------- async global->LDS path (gfx1250 ASYNCcnt engine) ----------
#if __has_builtin(__builtin_amdgcn_global_load_async_to_lds_b128) && \
    __has_builtin(__builtin_amdgcn_s_wait_asynccnt)
#define USE_ASYNC_LDS 1
#else
#define USE_ASYNC_LDS 0
#endif

#if USE_ASYNC_LDS
static __device__ __forceinline__ void async_copy16(const void* gsrc, void* ldst) {
  __builtin_amdgcn_global_load_async_to_lds_b128(
      (__attribute__((address_space(1))) v4i*)gsrc,
      (__attribute__((address_space(3))) v4i*)ldst,
      0, 0);
}
#endif

// =====================================================================
// 1) Segmented prefix scan of feat_map (C,T) -> cs stored (T+1, C):
//    cs[t*C + c] = sum_{u<t} feat[c, u]   (column t is contiguous)
// =====================================================================
constexpr int kSEG  = 256;
constexpr int kNSEG = kT / kSEG;   // 32

__global__ __launch_bounds__(256) void seg_sum_kernel(
    const float* __restrict__ feat, float* __restrict__ segsum) {
  int idx = blockIdx.x * blockDim.x + threadIdx.x;   // s*kC + c
  if (idx >= kNSEG * kC) return;
  int s = idx / kC;
  int c = idx % kC;
  const float* p = feat + (size_t)c * kT + (size_t)s * kSEG;
  float acc = 0.f;
  #pragma unroll 4
  for (int i = 0; i < kSEG; i += 4) {
    float4 v = *(const float4*)(p + i);
    acc += v.x + v.y + v.z + v.w;
  }
  segsum[idx] = acc;
}

__global__ __launch_bounds__(256) void seg_scan_kernel(float* __restrict__ segsum) {
  int c = blockIdx.x * blockDim.x + threadIdx.x;     // one thread per channel
  if (c >= kC) return;
  float acc = 0.f;
  for (int s = 0; s < kNSEG; ++s) {
    float v = segsum[(size_t)s * kC + c];
    segsum[(size_t)s * kC + c] = acc;                // exclusive
    acc += v;
  }
}

__global__ __launch_bounds__(256) void seg_write_kernel(
    const float* __restrict__ feat, const float* __restrict__ segsum,
    float* __restrict__ cs) {
  int idx = blockIdx.x * blockDim.x + threadIdx.x;   // s*kC + c
  if (idx >= kNSEG * kC) return;
  int s = idx / kC;
  int c = idx % kC;
  float acc = segsum[idx];
  if (s == 0) cs[c] = 0.f;                           // cs row t=0 is zero
  const float* p = feat + (size_t)c * kT + (size_t)s * kSEG;
  size_t ob = (size_t)(s * kSEG + 1) * kC + c;
  for (int i = 0; i < kSEG; ++i) {
    acc += p[i];
    cs[ob + (size_t)i * kC] = acc;                   // coalesced across c
  }
}

// =====================================================================
// 2) Boundary pooling -> e (N, 2C) in bf16
// =====================================================================
__global__ __launch_bounds__(128) void pool_kernel(
    const float* __restrict__ cs, const int* __restrict__ lp,
    const int* __restrict__ rp, __bf16* __restrict__ e) {
  int i = blockIdx.x;
  int li = lp[i], ri = rp[i];
  float w  = fmaxf((float)ri - (float)li, 1.0f);
  int   bw = max(1, (int)(0.15f * w));               // trunc toward zero (w>0)
  int ls = max(0, li - bw), le = min(kT, li + bw);
  int rs = max(0, ri - bw), re = min(kT, ri + bw);
  le = min(max(ls + 1, le), kT);
  re = min(max(rs + 1, re), kT);
  float invL = 1.0f / (float)(le - ls);
  float invR = 1.0f / (float)(re - rs);
  const float* cls = cs + (size_t)ls * kC;
  const float* cle = cs + (size_t)le * kC;
  const float* crs = cs + (size_t)rs * kC;
  const float* cre = cs + (size_t)re * kC;
  __bf16* eo = e + (size_t)i * k2C;
  for (int c = threadIdx.x; c < kC; c += blockDim.x) {
    eo[c]       = to_bf16((cle[c] - cls[c]) * invL);
    eo[c + kC]  = to_bf16((cre[c] - crs[c]) * invR);
  }
}

// =====================================================================
// 3) Transpose + convert W (rows x cols, fp32, row-major)
//    -> WT (cols x rows, bf16, row-major)  [WT[k*rows + j] = W[j*cols + k]]
// =====================================================================
__global__ __launch_bounds__(256) void wtrans_kernel(
    const float* __restrict__ W, __bf16* __restrict__ WT, int rows, int cols) {
  int idx = blockIdx.x * blockDim.x + threadIdx.x;
  if (idx >= rows * cols) return;
  int k = idx / rows;
  int j = idx % rows;
  WT[idx] = to_bf16(W[(size_t)j * cols + k]);
}

// =====================================================================
// 4) bf16 WMMA GEMM: OUT[m,n] = relu?(sum_k A[m,k]*B[k,n] + bias[n])
//    Block tile 128x128, BK=32, 256 threads = 8 waves (wave32),
//    wave grid 2x4, each wave: 4x2 tiles of v_wmma_f32_16x16x32_bf16.
//    Tile staging: async global->LDS (ASYNCcnt) with double buffering,
//    one workgroup barrier per K-step.
// =====================================================================
constexpr int BM = 128, BN = 128, BK = 32;

union V16U { uint4 u[2]; v16bf v; };

template<bool RELU, bool OUT_BF16>
__global__ __launch_bounds__(256) void gemm_bf16_kernel(
    const __bf16* __restrict__ A, const __bf16* __restrict__ B,
    const float* __restrict__ bias, void* __restrict__ Cout,
    int M, int Nn, int K) {
  __shared__ __bf16 As[2][BM * BK];   // [m][k] : byte(m,k) = m*64 + k*2
  __shared__ __bf16 Bs[2][BK * BN];   // [k][n] : byte(k,n) = k*256 + n*2

  const int tid   = threadIdx.x;
  const int lane  = tid & 31;
  const int wave  = tid >> 5;
  const int waveM = wave >> 2;        // 0..1  -> 64 rows each
  const int waveN = wave & 3;         // 0..3  -> 32 cols each
  const int rowBase = blockIdx.x * BM;
  const int colBase = blockIdx.y * BN;

  v8f acc[4][2];
  #pragma unroll
  for (int i = 0; i < 4; ++i)
    #pragma unroll
    for (int j = 0; j < 2; ++j)
      #pragma unroll
      for (int r = 0; r < 8; ++r) acc[i][j][r] = 0.f;

  const int halfsel = (lane < 16) ? 0 : 16;
  const int mrow    = lane & 15;

  // per-thread staging geometry: thread t owns bytes [t*32, t*32+32) of each tile
  const int lb   = tid * 32;
  const int arow = lb >> 6,  acb = lb & 63;    // A tile: 64B rows
  const int brow = lb >> 8,  bcb = lb & 255;   // B tile: 256B rows
  const char* aSrc0 = (const char*)A + (size_t)(rowBase + arow) * K * 2 + acb;
  const char* bSrc0 = (const char*)B + (size_t)colBase * 2 + bcb;

  const int nsteps = K / BK;

#if USE_ASYNC_LDS
  auto issue_tile = [&](int k0, int buf) {
    const char* as = aSrc0 + (size_t)k0 * 2;
    async_copy16(as,      (char*)As[buf] + lb);
    async_copy16(as + 16, (char*)As[buf] + lb + 16);
    const char* bs = bSrc0 + (size_t)(k0 + brow) * Nn * 2;
    async_copy16(bs,      (char*)Bs[buf] + lb);
    async_copy16(bs + 16, (char*)Bs[buf] + lb + 16);
  };
  issue_tile(0, 0);
#endif

  for (int i = 0; i < nsteps; ++i) {
    const int buf = i & 1;

#if USE_ASYNC_LDS
    if (i + 1 < nsteps) {
      issue_tile((i + 1) * BK, buf ^ 1);
      __builtin_amdgcn_s_wait_asynccnt(4);   // current tile's 4 copies done
    } else {
      __builtin_amdgcn_s_wait_asynccnt(0);
    }
    __syncthreads();                         // all threads' tile data visible
#else
    // synchronous fallback staging
    {
      const char* as = aSrc0 + (size_t)(i * BK) * 2;
      uint4* dst = (uint4*)((char*)As[buf] + lb);
      dst[0] = ((const uint4*)as)[0];
      dst[1] = ((const uint4*)as)[1];
      const char* bs = bSrc0 + (size_t)(i * BK + brow) * Nn * 2;
      uint4* dstb = (uint4*)((char*)Bs[buf] + lb);
      dstb[0] = ((const uint4*)bs)[0];
      dstb[1] = ((const uint4*)bs)[1];
    }
    __syncthreads();
#endif

    // --- A fragments: ISA 16-bit A 16x32 layout ---
    // lane<16: K = {0..7,16..23}; lane>=16: K = {8..15,24..31}
    v16bf afrag[4];
    #pragma unroll
    for (int mt = 0; mt < 4; ++mt) {
      const char* base = (const char*)As[buf] +
          (waveM * 64 + mt * 16 + mrow) * 64 + halfsel;
      V16U t;
      t.u[0] = *(const uint4*)(base);
      t.u[1] = *(const uint4*)(base + 32);
      afrag[mt] = t.v;
    }
    // --- B fragments: N = lane%16; K order same as A (table order) ---
    v16bf bfrag[2];
    #pragma unroll
    for (int nt = 0; nt < 2; ++nt) {
      const char* base = (const char*)Bs[buf] +
          (halfsel / 2) * 256 + (waveN * 32 + nt * 16 + mrow) * 2;
      V16U t;
      #pragma unroll
      for (int s = 0; s < 16; ++s)
        t.v[s] = *(const __bf16*)(base + (s < 8 ? s : s + 8) * 256);
      bfrag[nt] = t.v;
    }

    #pragma unroll
    for (int mt = 0; mt < 4; ++mt)
      #pragma unroll
      for (int nt = 0; nt < 2; ++nt)
        acc[mt][nt] = __builtin_amdgcn_wmma_f32_16x16x32_bf16(
            false, afrag[mt], false, bfrag[nt],
            (short)0, acc[mt][nt], false, false);

#if !USE_ASYNC_LDS
    __syncthreads();   // single-barrier double buffering only valid w/ async
#endif
  }

  // --- epilogue: D layout -> lanes 0-15: M=r, lanes 16-31: M=r+8; N=lane%16
  const int mofs = (lane < 16) ? 0 : 8;
  #pragma unroll
  for (int mt = 0; mt < 4; ++mt) {
    #pragma unroll
    for (int nt = 0; nt < 2; ++nt) {
      int n = colBase + waveN * 32 + nt * 16 + (lane & 15);
      float bz = bias[n];
      #pragma unroll
      for (int r = 0; r < 8; ++r) {
        int m = rowBase + waveM * 64 + mt * 16 + r + mofs;
        float v = acc[mt][nt][r] + bz;
        if (RELU) v = fmaxf(v, 0.f);
        if (OUT_BF16)
          ((__bf16*)Cout)[(size_t)m * Nn + n] = to_bf16(v);
        else
          ((float*)Cout)[(size_t)m * Nn + n] = v;
      }
    }
  }
}

// =====================================================================
// Launch
// =====================================================================
extern "C" void kernel_launch(void* const* d_in, const int* in_sizes, int n_in,
                              void* d_out, int out_size, void* d_ws, size_t ws_size,
                              hipStream_t stream) {
  const float* feat = (const float*)d_in[0];   // (C,T)
  const int*   lp   = (const int*)d_in[1];     // (N,)
  const int*   rp   = (const int*)d_in[2];     // (N,)
  // d_in[3] = T scalar (known = 8192)
  const float* W1   = (const float*)d_in[4];   // (HID, 2C)
  const float* b1   = (const float*)d_in[5];   // (HID,)
  const float* W2   = (const float*)d_in[6];   // (OUT, HID)
  const float* b2   = (const float*)d_in[7];   // (OUT,)

  // ---- workspace partition (256B aligned) ----
  char* ws = (char*)d_ws;
  size_t off = 0;
  auto alloc = [&](size_t bytes) {
    char* p = ws + off;
    off += (bytes + 255) & ~(size_t)255;
    return p;
  };
  float*  cs     = (float*) alloc((size_t)(kT + 1) * kC * sizeof(float)); // 16.8MB
  __bf16* e      = (__bf16*)alloc((size_t)kN * k2C * sizeof(__bf16));     // 32MB
  __bf16* h      = (__bf16*)alloc((size_t)kN * kHID * sizeof(__bf16));    // 32MB
  __bf16* w1t    = (__bf16*)alloc((size_t)k2C * kHID * sizeof(__bf16));   // 2MB
  __bf16* w2t    = (__bf16*)alloc((size_t)kHID * kOUT * sizeof(__bf16));  // 1MB
  float*  segsum = (float*) alloc((size_t)kNSEG * kC * sizeof(float));    // 64KB

  // ---- 1) prefix scan (3-phase) ----
  {
    int tot = kNSEG * kC;                       // 16384
    seg_sum_kernel <<<(tot + 255) / 256, 256, 0, stream>>>(feat, segsum);
    seg_scan_kernel<<<(kC + 255) / 256, 256, 0, stream>>>(segsum);
    seg_write_kernel<<<(tot + 255) / 256, 256, 0, stream>>>(feat, segsum, cs);
  }

  // ---- 2) weight transpose + bf16 convert ----
  wtrans_kernel<<<(kHID * k2C + 255) / 256, 256, 0, stream>>>(W1, w1t, kHID, k2C);
  wtrans_kernel<<<(kOUT * kHID + 255) / 256, 256, 0, stream>>>(W2, w2t, kOUT, kHID);

  // ---- 3) boundary pooling -> e (bf16) ----
  pool_kernel<<<kN, 128, 0, stream>>>(cs, lp, rp, e);

  // ---- 4) GEMM1: h = relu(e @ w1t + b1)  (16384x1024 @ 1024x1024) ----
  {
    dim3 grid(kN / BM, kHID / BN);
    gemm_bf16_kernel<true, true><<<grid, 256, 0, stream>>>(
        e, w1t, b1, (void*)h, kN, kHID, k2C);
  }
  // ---- 5) GEMM2: out = h @ w2t + b2  (16384x1024 @ 1024x512, fp32 out) ----
  {
    dim3 grid(kN / BM, kOUT / BN);
    gemm_bf16_kernel<false, false><<<grid, 256, 0, stream>>>(
        h, w2t, b2, d_out, kN, kOUT, kHID);
  }
}
